// VectorNetEncoder_51582557225723
// MI455X (gfx1250) — matrix operand order repaired and checked by reading, not compiled, not run on hardware
//
#include <hip/hip_runtime.h>
#include <hip/hip_bf16.h>

// ---------------------------------------------------------------------------
// VectorNet encoder for MI455X (gfx1250, wave32, WMMA f16 -> f32 accumulate)
// ---------------------------------------------------------------------------

typedef __attribute__((ext_vector_type(16))) _Float16 v16h;
typedef __attribute__((ext_vector_type(8)))  _Float16 v8h;
typedef __attribute__((ext_vector_type(8)))  float    v8f;

#define N_OBJ   150
#define BATCH   512
#define TRAJ    76800            // BATCH * N_OBJ
#define TB      16               // trajectories per workgroup (kernel 1)
#define ROWS    144              // TB * 9 steps
#define SRA0    40               // padded stride (halves) for 9x10 input, K padded to 32
#define SRA1    136              // padded stride (halves) for 128-wide activations
#define SRB1    136              // padded W^T stride, K=128 layers
#define SRS     168              // padded stride (halves) kernel 2, K padded to 160

// ---------------------------------------------------------------------------
// Generic WMMA tile GEMM: C[MT*16 x 128] = A[MT*16 x KS*32] * W^T-rows
//   A: LDS f16, row stride sra (halves).  A-frag per ISA 16-bit A 16x32 layout.
//   WT: LDS f16, row n holds K values of output column n, stride srb (halves).
//   Cb: LDS f32, row stride 128.  8 waves, wave w owns output-column tile w.
// ---------------------------------------------------------------------------
template<int MT, int KS>
__device__ inline void wmma_gemm(const _Float16* A, int sra,
                                 const _Float16* WT, int srb,
                                 float* Cb, int tid) {
  const int lane = tid & 31;
  const int nt   = tid >> 5;            // 8 waves -> 8 N tiles
  const int nIdx = lane & 15;
  const int kbB  = (lane >> 4) * 8;     // B frag: K kb..kb+7 then kb+16..kb+23
  v16h bfrag[KS];
#pragma unroll
  for (int ks = 0; ks < KS; ++ks) {
    const _Float16* p = WT + (nt * 16 + nIdx) * srb + ks * 32 + kbB;
    ((v8h*)&bfrag[ks])[0] = *(const v8h*)(p);
    ((v8h*)&bfrag[ks])[1] = *(const v8h*)(p + 16);
  }
  const int mr  = lane & 15;
  const int kbA = (lane >> 4) * 8;      // A frag: K kb..kb+7 then kb+16..kb+23
#pragma unroll
  for (int mt = 0; mt < MT; ++mt) {
    v8f c = {};
#pragma unroll
    for (int ks = 0; ks < KS; ++ks) {
      v16h a;
      const _Float16* p = A + (mt * 16 + mr) * sra + ks * 32 + kbA;
      ((v8h*)&a)[0] = *(const v8h*)(p);
      ((v8h*)&a)[1] = *(const v8h*)(p + 16);
      c = __builtin_amdgcn_wmma_f32_16x16x32_f16(false, a, false, bfrag[ks],
                                                 (short)0, c, false, false);
    }
    const int r0 = mt * 16 + (lane >> 4) * 8;
    const int cc = nt * 16 + nIdx;
#pragma unroll
    for (int r = 0; r < 8; ++r) Cb[(r0 + r) * 128 + cc] = c[r];
  }
}

__device__ inline void copy16(const _Float16* src, _Float16* dst, int n16, int tid) {
  const uint4* s4 = (const uint4*)src;
  uint4* d4 = (uint4*)dst;
  for (int i = tid; i < n16; i += 256) d4[i] = s4[i];
}

// ---------------------------------------------------------------------------
// Kernel 0: weight prep.  Builds f16 W^T layouts (padded strides, zero-padded
// K) and folded biases  eb = b + relu(agg.b2) * colsum(W[128:256,:]).
// ---------------------------------------------------------------------------
struct PrepP {
  const float* w0l1;          // glp0.l1.w   (10 x 128)
  const float* src128[6];     // glp0l2, glp1l1, glp1l2, glp2l1, glp2l2, final_linear
  const float* lb[3];         // glp1.l1.b, glp2.l1.b, final_linear.b
  const float* lw[3];         // glp1.l1.w, glp2.l1.w, final_linear.w (256x128)
  const float* aggb2[3];      // agg{0,1,2}.b2 (scalar)
  const float* kw;            // attn.k.w (130x128)
  const float* vw;            // attn.v.w (130x128)
  _Float16* wt0;              // [128][SRA0]
  _Float16* wtA;              // 6 x [128][SRB1]
  float*    ebias;            // 3 x 128
  _Float16* wtK;              // [128][SRS]
  _Float16* wtV;              // [128][SRS]
};

__global__ __launch_bounds__(256) void prep_kernel(PrepP p) {
  const int blk = blockIdx.x, tid = threadIdx.x;
  if (blk == 0) {
    for (int i = tid; i < 128 * SRA0; i += 256) {
      int n = i / SRA0, k = i % SRA0;
      p.wt0[i] = (_Float16)((k < 10) ? p.w0l1[k * 128 + n] : 0.f);
    }
  } else if (blk <= 6) {
    const float* src = p.src128[blk - 1];
    _Float16* dst = p.wtA + (size_t)(blk - 1) * 128 * SRB1;
    for (int i = tid; i < 128 * SRB1; i += 256) {
      int n = i / SRB1, k = i % SRB1;
      dst[i] = (_Float16)((k < 128) ? src[k * 128 + n] : 0.f);
    }
  } else if (blk == 7 || blk == 8) {
    const float* src = (blk == 7) ? p.kw : p.vw;
    _Float16* dst = (blk == 7) ? p.wtK : p.wtV;
    for (int i = tid; i < 128 * SRS; i += 256) {
      int n = i / SRS, k = i % SRS;
      dst[i] = (_Float16)((k < 130) ? src[k * 128 + n] : 0.f);
    }
  } else {
    for (int i = tid; i < 3 * 128; i += 256) {
      int r = i >> 7, n = i & 127;
      float c = fmaxf(p.aggb2[r][0], 0.f);   // LN over size-1 axis == shift b2
      float s = 0.f;
      for (int k = 128; k < 256; ++k) s += p.lw[r][k * 128 + n];
      p.ebias[i] = p.lb[r][n] + c * s;
    }
  }
}

// ---------------------------------------------------------------------------
// Kernel 1: subgraph.  One WG = 16 trajectories = 144 GEMM rows.
// 7 linears (L0 K=10->pad32, L1..L6 K=128), LN+ReLU after L0..L5,
// then step-aggregation (final_agg) + L2-normalize, emit sub[m][130].
// ---------------------------------------------------------------------------
struct SubP {
  const float* obs;
  const _Float16* wt[7];
  const float* bias[7];       // [2],[4],[6] are folded ebias rows
  const float* gamma[6];
  const float* beta[6];
  const float* faw;           // final_agg.w (9)
  const float* fab;           // final_agg.b (1)
  float* sub;                 // out [76800][130]
};

__device__ inline void ln_relu(const float* C, _Float16* H, const float* bias,
                               const float* g, const float* bt, int tid) {
  const int grp = tid >> 4, li = tid & 15;
  for (int r = grp; r < ROWS; r += 16) {
    float x[8]; float s = 0.f, s2 = 0.f;
#pragma unroll
    for (int ci = 0; ci < 8; ++ci) {
      int c = li + ci * 16;
      float v = C[r * 128 + c] + bias[c];
      x[ci] = v; s += v; s2 += v * v;
    }
#pragma unroll
    for (int off = 1; off < 16; off <<= 1) {
      s += __shfl_xor(s, off, 32);
      s2 += __shfl_xor(s2, off, 32);
    }
    float mu = s * (1.f / 128.f);
    float var = s2 * (1.f / 128.f) - mu * mu;
    float rs = rsqrtf(var + 1e-5f);
#pragma unroll
    for (int ci = 0; ci < 8; ++ci) {
      int c = li + ci * 16;
      float v = (x[ci] - mu) * rs * g[c] + bt[c];
      H[r * SRA1 + c] = (_Float16)fmaxf(v, 0.f);
    }
  }
}

__global__ __launch_bounds__(256) void subgraph_kernel(SubP p) {
  extern __shared__ char smem[];
  _Float16* X0 = (_Float16*)(smem);                      // 144*40*2   = 11520
  _Float16* W  = (_Float16*)(smem + 11520);              // 128*136*2  = 34816
  _Float16* H  = (_Float16*)(smem + 11520 + 34816);      // 144*136*2  = 39168
  float*    C  = (float*)   (smem + 11520 + 34816 + 39168); // 144*128*4 = 73728
  const int wg = blockIdx.x, tid = threadIdx.x;

  // stage raw 9x10 trajectory features (K padded to 32 with zeros)
  for (int i = tid; i < ROWS * SRA0; i += 256) {
    int lr = i / SRA0, f = i % SRA0;
    int t = lr / 9, s = lr % 9;
    int m = wg * TB + t;
    int b = m / N_OBJ, o = m % N_OBJ;
    float v = (f < 10) ? p.obs[((b * 151 + o) * 9 + s) * 10 + f] : 0.f;
    X0[i] = (_Float16)v;
  }
  copy16(p.wt[0], W, 128 * SRA0 * 2 / 16, tid);
  __syncthreads();

  wmma_gemm<9, 1>(X0, SRA0, W, SRA0, C, tid);
  __syncthreads();
  ln_relu(C, H, p.bias[0], p.gamma[0], p.beta[0], tid);
  __syncthreads();

  for (int L = 1; L <= 5; ++L) {
    copy16(p.wt[L], W, 128 * SRB1 * 2 / 16, tid);
    __syncthreads();
    wmma_gemm<9, 4>(H, SRA1, W, SRB1, C, tid);
    __syncthreads();
    ln_relu(C, H, p.bias[L], p.gamma[L], p.beta[L], tid);
    __syncthreads();
  }

  // final linear (no LN)
  copy16(p.wt[6], W, 128 * SRB1 * 2 / 16, tid);
  __syncthreads();
  wmma_gemm<9, 4>(H, SRA1, W, SRB1, C, tid);
  __syncthreads();

  // final_agg over the 9 steps + L2 normalize, append rel_pos
  {
    const int g = tid >> 4, li = tid & 15;
    const int m = wg * TB + g;
    const int b = m / N_OBJ, o = m % N_OBJ;
    const float* eb = p.bias[6];
    const float bfa = p.fab[0];
    float wfa[9];
#pragma unroll
    for (int s = 0; s < 9; ++s) wfa[s] = p.faw[s];
    float z[8]; float ss = 0.f;
#pragma unroll
    for (int ci = 0; ci < 8; ++ci) {
      int c = li + ci * 16;
      float ebc = eb[c];
      float acc = bfa;
#pragma unroll
      for (int s = 0; s < 9; ++s) acc += (C[(g * 9 + s) * 128 + c] + ebc) * wfa[s];
      z[ci] = acc; ss += acc * acc;
    }
#pragma unroll
    for (int off = 1; off < 16; off <<= 1) ss += __shfl_xor(ss, off, 32);
    float scale = 1.f / fmaxf(sqrtf(ss), 1e-12f);
    float* srow = p.sub + (size_t)m * 130;
#pragma unroll
    for (int ci = 0; ci < 8; ++ci) srow[li + ci * 16] = z[ci] * scale;
    if (li == 0) {
      srow[128] = p.obs[((b * 151 + o) * 9 + 8) * 10 + 0];
      srow[129] = p.obs[((b * 151 + o) * 9 + 8) * 10 + 1];
    }
  }
}

// ---------------------------------------------------------------------------
// Kernel 2: ego-only attention + projection.  One WG per batch element.
// K = S*Wk (bias drops out of softmax), V = S*Wv (bias added after prob-sum).
// ---------------------------------------------------------------------------
struct AttnP {
  const float* sub; const float* obs;
  const _Float16* wtK; const _Float16* wtV;
  const float* qw; const float* qb; const float* vb;
  const float* pw; const float* plb; const float* pg; const float* pbb;
  float* out;
};

__device__ inline float blk_sum(float v, float* red, int tid) {
#pragma unroll
  for (int off = 16; off > 0; off >>= 1) v += __shfl_xor(v, off, 32);
  __syncthreads();
  if ((tid & 31) == 0) red[tid >> 5] = v;
  __syncthreads();
  float t = 0.f;
#pragma unroll
  for (int i = 0; i < 8; ++i) t += red[i];
  return t;
}

__device__ inline float blk_max(float v, float* red, int tid) {
#pragma unroll
  for (int off = 16; off > 0; off >>= 1) v = fmaxf(v, __shfl_xor(v, off, 32));
  __syncthreads();
  if ((tid & 31) == 0) red[tid >> 5] = v;
  __syncthreads();
  float t = red[0];
#pragma unroll
  for (int i = 1; i < 8; ++i) t = fmaxf(t, red[i]);
  return t;
}

__global__ __launch_bounds__(256) void attn_proj_kernel(AttnP p) {
  extern __shared__ char smem[];
  _Float16* S  = (_Float16*)(smem);                  // 160*168*2 = 53760
  _Float16* WT = (_Float16*)(smem + 53760);          // 128*168*2 = 43008
  float*    KV = (float*)(smem + 53760 + 43008);     // 160*128*4 = 81920
  float*    qv = (float*)(smem + 178688);            // 128
  float*    sc = qv + 128;                           // 160
  float*    red = sc + 160;                          // 32
  float*    comb = red + 32;                         // 132
  const int b = blockIdx.x, tid = threadIdx.x;
  const size_t sb = (size_t)b * N_OBJ * 130;

  for (int i = tid; i < 160 * SRS; i += 256) {
    int r = i / SRS, c = i % SRS;
    float v = (r < N_OBJ && c < 130) ? p.sub[sb + r * 130 + c] : 0.f;
    S[i] = (_Float16)v;
  }
  copy16(p.wtK, WT, 128 * SRS * 2 / 16, tid);
  __syncthreads();

  wmma_gemm<10, 5>(S, SRS, WT, SRS, KV, tid);        // K matrix
  __syncthreads();

  if (tid < 128) {                                   // ego query (f32 exact)
    float acc = p.qb[tid];
    for (int k = 0; k < 130; ++k) acc += p.sub[sb + k] * p.qw[k * 128 + tid];
    qv[tid] = acc;
  }
  __syncthreads();

  float s_j = -1e30f;
  if (tid < N_OBJ) {
    float acc = 0.f;
    for (int c = 0; c < 128; ++c) acc += qv[c] * KV[tid * 128 + c];
    s_j = acc * 0.08838834764831845f;                // 1/sqrt(128)
  }
  float mx = blk_max(s_j, red, tid);
  float e = (tid < N_OBJ) ? __expf(s_j - mx) : 0.f;
  float ssum = blk_sum(e, red, tid);
  if (tid < 160) sc[tid] = (tid < N_OBJ) ? (e / ssum) : 0.f;
  __syncthreads();

  copy16(p.wtV, WT, 128 * SRS * 2 / 16, tid);
  __syncthreads();
  wmma_gemm<10, 5>(S, SRS, WT, SRS, KV, tid);        // V matrix
  __syncthreads();

  if (tid < 128) {
    float acc = p.vb[tid];
    for (int j = 0; j < N_OBJ; ++j) acc += sc[j] * KV[j * 128 + tid];
    comb[tid] = acc;
  } else if (tid == 128) {
    comb[128] = p.obs[((b * 151 + 150) * 9 + 8) * 10 + 0];
  } else if (tid == 129) {
    comb[129] = p.obs[((b * 151 + 150) * 9 + 8) * 10 + 1];
  }
  __syncthreads();

  // projection 130 -> 512 + LayerNorm + ReLU
  float t0 = p.plb[tid], t1 = p.plb[tid + 256];
  for (int k = 0; k < 130; ++k) {
    float ck = comb[k];
    t0 += ck * p.pw[k * 512 + tid];
    t1 += ck * p.pw[k * 512 + tid + 256];
  }
  float s1 = blk_sum(t0 + t1, red, tid);
  float s2 = blk_sum(t0 * t0 + t1 * t1, red, tid);
  float mu = s1 * (1.f / 512.f);
  float var = s2 * (1.f / 512.f) - mu * mu;
  float rs = rsqrtf(var + 1e-5f);
  float* o = p.out + (size_t)b * 512;
  o[tid]       = fmaxf((t0 - mu) * rs * p.pg[tid] + p.pbb[tid], 0.f);
  o[tid + 256] = fmaxf((t1 - mu) * rs * p.pg[tid + 256] + p.pbb[tid + 256], 0.f);
}

// ---------------------------------------------------------------------------
// Host launch.  Input order = jax pytree (sorted dict keys):
//  0 obs | 1 k.b 2 k.w 3 q.b 4 q.w 5 v.b 6 v.w | 7 proj.b 8 proj.g 9 lin.b
//  10 lin.w | agg0: 11 b1 12 b2 13 g1 14 g2 15 l1.b 16 l1.w 17 l2.b 18 l2.w
//  agg1: 19..26  agg2: 27..34 | 35 fa.b 36 fa.w | 37 fl.b 38 fl.w |
//  glp0: 39 b1 40 b2 41 g1 42 g2 43 l1.b 44 l1.w 45 l2.b 46 l2.w
//  glp1: 47..54  glp2: 55..62
// ---------------------------------------------------------------------------
extern "C" void kernel_launch(void* const* d_in, const int* in_sizes, int n_in,
                              void* d_out, int out_size, void* d_ws, size_t ws_size,
                              hipStream_t stream) {
  (void)in_sizes; (void)n_in; (void)out_size; (void)ws_size;
  const float* in[64];
  for (int i = 0; i < 63; ++i) in[i] = (const float*)d_in[i];

  char* ws = (char*)d_ws;
  float*    sub  = (float*)ws;                                  // 39,936,000 B
  _Float16* wt0  = (_Float16*)(ws + 39936000);                  // 10,240 B
  _Float16* wtA  = (_Float16*)(ws + 39936000 + 10240);          // 6 * 34,816 B
  float*    eb   = (float*)(ws + 39936000 + 10240 + 208896);    // 1,536 B
  _Float16* wtK  = (_Float16*)(ws + 40156672);                  // 43,008 B
  _Float16* wtV  = (_Float16*)(ws + 40199680);                  // 43,008 B

  PrepP pp;
  pp.w0l1 = in[44];
  pp.src128[0] = in[46]; pp.src128[1] = in[52]; pp.src128[2] = in[54];
  pp.src128[3] = in[60]; pp.src128[4] = in[62]; pp.src128[5] = in[38];
  pp.lb[0] = in[51]; pp.lb[1] = in[59]; pp.lb[2] = in[37];
  pp.lw[0] = in[52]; pp.lw[1] = in[60]; pp.lw[2] = in[38];
  pp.aggb2[0] = in[12]; pp.aggb2[1] = in[20]; pp.aggb2[2] = in[28];
  pp.kw = in[2]; pp.vw = in[6];
  pp.wt0 = wt0; pp.wtA = wtA; pp.ebias = eb; pp.wtK = wtK; pp.wtV = wtV;
  prep_kernel<<<10, 256, 0, stream>>>(pp);

  SubP sp;
  sp.obs = in[0];
  sp.wt[0] = wt0;
  for (int L = 1; L <= 6; ++L) sp.wt[L] = wtA + (size_t)(L - 1) * 128 * SRB1;
  sp.bias[0] = in[43]; sp.bias[1] = in[45]; sp.bias[2] = eb;
  sp.bias[3] = in[53]; sp.bias[4] = eb + 128; sp.bias[5] = in[61];
  sp.bias[6] = eb + 256;
  sp.gamma[0] = in[41]; sp.gamma[1] = in[42]; sp.gamma[2] = in[49];
  sp.gamma[3] = in[50]; sp.gamma[4] = in[57]; sp.gamma[5] = in[58];
  sp.beta[0]  = in[39]; sp.beta[1]  = in[40]; sp.beta[2]  = in[47];
  sp.beta[3]  = in[48]; sp.beta[4]  = in[55]; sp.beta[5]  = in[56];
  sp.faw = in[36]; sp.fab = in[35];
  sp.sub = sub;
  const int smem1 = 11520 + 34816 + 39168 + 73728;              // 159,232 B
  subgraph_kernel<<<TRAJ / TB, 256, smem1, stream>>>(sp);

  AttnP ap;
  ap.sub = sub; ap.obs = in[0];
  ap.wtK = wtK; ap.wtV = wtV;
  ap.qw = in[4]; ap.qb = in[3]; ap.vb = in[5];
  ap.pw = in[10]; ap.plb = in[9]; ap.pg = in[8]; ap.pbb = in[7];
  ap.out = (float*)d_out;
  const int smem2 = 178688 + (128 + 160 + 32 + 132) * 4;        // 180,496 B
  attn_proj_kernel<<<BATCH, 256, smem2, stream>>>(ap);
}
